// TTSModel_84447646973978
// MI455X (gfx1250) — matrix operand order repaired and checked by reading, not compiled, not run on hardware
//
#include <hip/hip_runtime.h>
#include <hip/hip_bf16.h>

// ---------------------------------------------------------------------------
// Types for CDNA5 WMMA bf16 (16x16x32, f32 accum)
// ---------------------------------------------------------------------------
typedef __attribute__((ext_vector_type(16))) __bf16 v16bf;
typedef __attribute__((ext_vector_type(8)))  float  v8f;

// Model dims
#define Bsz   16
#define Tlen  128
#define Hdim  256
#define Edim  512
#define MELD  512
#define NHead 8
#define HDd   64
#define NSTEP 1000

// ---------------------------------------------------------------------------
// Fragment loaders (bf16, 16x16x32 WMMA)
// A layout (16x32, MxK): lanes 0-15 row M=lane, K = {kb+0..7, kb+16..23};
//                        lanes 16-31 same rows, K offset +8 in each half.
// B layout (32x16, KxN): B(k,n)=W[n,k] (W row-major [N,K]) ->
//                        lanes 0-15 col N=lane, K = {kb+0..7, kb+16..23};
//                        lanes 16-31 K offset +8 in each half (same as A).
// Both are two contiguous 16-byte loads per lane.
// ---------------------------------------------------------------------------
__device__ __forceinline__ v16bf load_frag_a(const __hip_bfloat16* rowp, int kb, int lane) {
    const __hip_bfloat16* p = rowp + kb + (((lane >> 4) & 1) << 3);
    union { v16bf v; uint4 q[2]; } f;
    f.q[0] = *(const uint4*)(p);
    f.q[1] = *(const uint4*)(p + 16);
    return f.v;
}

__device__ __forceinline__ v16bf load_frag_b(const __hip_bfloat16* rowp, int kb, int lane) {
    const __hip_bfloat16* p = rowp + kb + (((lane >> 4) & 1) << 3);
    union { v16bf v; uint4 q[2]; } f;
    f.q[0] = *(const uint4*)(p);
    f.q[1] = *(const uint4*)(p + 16);
    return f.v;
}

// One 16x16 output tile: acc += A[mbase:mbase+16, 0:K] * W[nbase:nbase+16, 0:K]^T
__device__ __forceinline__ v8f mm_acc(const __hip_bfloat16* A, int lda, int mbase,
                                      const __hip_bfloat16* W, int ldw, int nbase,
                                      int K, v8f acc) {
    const int lane = threadIdx.x & 31;
    const __hip_bfloat16* arow = A + (size_t)(mbase + (lane & 15)) * lda;
    const __hip_bfloat16* wrow = W + (size_t)(nbase + (lane & 15)) * ldw;
    for (int kb = 0; kb < K; kb += 32) {
        v16bf a = load_frag_a(arow, kb, lane);
        v16bf b = load_frag_b(wrow, kb, lane);
        acc = __builtin_amdgcn_wmma_f32_16x16x32_bf16(false, a, false, b,
                                                      (short)0, acc, false, false);
    }
    return acc;
}

__device__ __forceinline__ float sigm(float x)  { return 1.0f / (1.0f + __expf(-x)); }
__device__ __forceinline__ float tanhfast(float x) { return 2.0f * sigm(2.0f * x) - 1.0f; }

// ---------------------------------------------------------------------------
// K0: f32 -> bf16 weight conversion
// ---------------------------------------------------------------------------
__global__ void cvt_bf16_kernel(const float* __restrict__ src,
                                __hip_bfloat16* __restrict__ dst, int n) {
    int i = blockIdx.x * blockDim.x + threadIdx.x;
    if (i < n) dst[i] = __float2bfloat16(src[i]);
}

// ---------------------------------------------------------------------------
// K1: embedding -> x_bf [B*T, H] bf16  (row m = b*T + t)
// ---------------------------------------------------------------------------
__global__ void embed_kernel(const int* __restrict__ tok, const int* __restrict__ spk,
                             const float* __restrict__ et, const float* __restrict__ es,
                             __hip_bfloat16* __restrict__ xbf) {
    int i = blockIdx.x * 256 + threadIdx.x;        // over B*T*H
    int hcol = i & (Hdim - 1);
    int m    = i >> 8;                             // b*T + t
    int b    = m >> 7;
    float v  = et[(size_t)tok[m] * Hdim + hcol] + es[(size_t)spk[b] * Hdim + hcol];
    xbf[i] = __float2bfloat16(v);
}

// ---------------------------------------------------------------------------
// K2: generic GEMM  C[M,N] f32 = A[M,K]bf16 @ W[N,K]^T + b1 (+ b2)
//     grid.x = M/16 (m-tile), grid.y = N/128, block 256 (8 waves = 8 n-tiles)
// ---------------------------------------------------------------------------
__global__ void __launch_bounds__(256)
gemm_bias_f32_kernel(const __hip_bfloat16* __restrict__ A,
                     const __hip_bfloat16* __restrict__ W,
                     const float* __restrict__ b1, const float* __restrict__ b2,
                     float* __restrict__ C, int K, int N) {
    int mt = blockIdx.x;
    int nt = blockIdx.y * 8 + (threadIdx.x >> 5);
    v8f acc = {};
    acc = mm_acc(A, K, mt * 16, W, K, nt * 16, K, acc);
    int lane = threadIdx.x & 31;
    int n = nt * 16 + (lane & 15);
    float bias = b1[n];
    if (b2) bias += b2[n];
    int mo = (lane >> 4) * 8;
    for (int r = 0; r < 8; ++r)
        C[(size_t)(mt * 16 + mo + r) * N + n] = acc[r] + bias;
}

// ---------------------------------------------------------------------------
// K3: K/V projection: enc[2048,512]bf16 @ W[512,512]^T + b -> f32 [B,NH,T,HD]
//     (f32 so the decode attention streams float4 from L2 with no unpack)
// ---------------------------------------------------------------------------
__global__ void __launch_bounds__(256)
gemm_kv_kernel(const __hip_bfloat16* __restrict__ enc,
               const __hip_bfloat16* __restrict__ W,
               const float* __restrict__ bias,
               float* __restrict__ out) {
    int mt = blockIdx.x;
    int nt = blockIdx.y * 8 + (threadIdx.x >> 5);
    v8f acc = {};
    acc = mm_acc(enc, Edim, mt * 16, W, Edim, nt * 16, Edim, acc);
    int lane = threadIdx.x & 31;
    int n = nt * 16 + (lane & 15);
    int h = n >> 6, d = n & 63;
    int mo = (lane >> 4) * 8;
    for (int r = 0; r < 8; ++r) {
        int m = mt * 16 + mo + r;        // b*T + t
        int b = m >> 7, t = m & 127;
        out[(((size_t)(b * NHead + h) * Tlen + t) * HDd) + d] = acc[r] + bias[n];
    }
}

// ---------------------------------------------------------------------------
// K4: bidirectional encoder LSTM (grid.x = 2 -> direction), persistent block,
//     512 threads = 16 wave32 (4 per SIMD32) for latency hiding.
// ---------------------------------------------------------------------------
__global__ void __launch_bounds__(512)
lstm_kernel(const float* __restrict__ xpre_f, const float* __restrict__ xpre_b,
            const __hip_bfloat16* __restrict__ Whh_f,
            const __hip_bfloat16* __restrict__ Whh_b,
            __hip_bfloat16* __restrict__ encbf /* [2048,512] */) {
    int dir = blockIdx.x;
    const float* xpre           = dir ? xpre_b : xpre_f;
    const __hip_bfloat16* Whh   = dir ? Whh_b : Whh_f;

    __shared__ alignas(16) __hip_bfloat16 s_h[Bsz][Hdim];    // 8 KB
    __shared__ alignas(16) float          s_c[Bsz][Hdim];    // 16 KB
    __shared__ alignas(16) float          s_g[Bsz][4 * Hdim];// 64 KB

    int tid = threadIdx.x, lane = tid & 31, w = tid >> 5;
    __hip_bfloat16 bz = __float2bfloat16(0.0f);
    for (int i = tid; i < Bsz * Hdim; i += 512) {
        ((__hip_bfloat16*)s_h)[i] = bz;
        ((float*)s_c)[i] = 0.0f;
    }
    __syncthreads();

    for (int step = 0; step < Tlen; ++step) {
        int t = dir ? (Tlen - 1 - step) : step;
        // gates = h @ Whh^T  (+ precomputed x@Wih^T + bih + bhh)
        for (int nt = w; nt < 64; nt += 16) {
            v8f acc = {};
            acc = mm_acc(&s_h[0][0], Hdim, 0, Whh, Hdim, nt * 16, Hdim, acc);
            int n = nt * 16 + (lane & 15);
            int mo = (lane >> 4) * 8;
            for (int r = 0; r < 8; ++r) {
                int b = mo + r;
                s_g[b][n] = acc[r] + xpre[((size_t)(b * Tlen + t)) * 1024 + n];
            }
        }
        __syncthreads();
        for (int i = tid; i < Bsz * Hdim; i += 512) {
            int b = i >> 8, j = i & (Hdim - 1);
            float ig = sigm(s_g[b][j]);
            float fg = sigm(s_g[b][Hdim + j]);
            float gg = tanhfast(s_g[b][2 * Hdim + j]);
            float og = sigm(s_g[b][3 * Hdim + j]);
            float c  = fg * s_c[b][j] + ig * gg;
            float h  = og * tanhfast(c);
            s_c[b][j] = c;
            s_h[b][j] = __float2bfloat16(h);
            encbf[((size_t)(b * Tlen + t)) * Edim + dir * Hdim + j] = __float2bfloat16(h);
        }
        __syncthreads();
    }
}

// ---------------------------------------------------------------------------
// K5: persistent decode loop (1 block, 16 wave32, 1000 steps)
// LDS pool with non-overlapping lifetimes:
//   region A (32 KB): q f32 [16][512] (phases 1-2)  -> gates bf16 [16][1024] (4-5)
//   region B (65 KB): softmax scores, stride 130, col map t+(t>>6)
//   region C (16 KB): ctx bf16 [16][512] (phases 2-3)
// ---------------------------------------------------------------------------
#define POOL_A 0
#define POOL_B 32768
#define POOL_C (32768 + 130 * 128 * 4)
#define POOL_SZ (POOL_C + 16 * 512 * 2)
#define SC(buf, row, t) (buf)[(row) * 130 + (t) + ((t) >> 6)]

__global__ void __launch_bounds__(512)
decode_kernel(const __hip_bfloat16* __restrict__ Wq,   const float* __restrict__ bq,
              const __hip_bfloat16* __restrict__ Wo,   const float* __restrict__ bo,
              const __hip_bfloat16* __restrict__ Wih,  const float* __restrict__ bih,
              const float* __restrict__ bhh,
              const __hip_bfloat16* __restrict__ Wmel, const float* __restrict__ bmel,
              const __hip_bfloat16* __restrict__ Wstop,const float* __restrict__ bstop,
              const float* __restrict__ Kp,            const float* __restrict__ Vp,
              float* __restrict__ mel_out, float* __restrict__ stop_out,
              float* __restrict__ attn_out) {
    __shared__ alignas(16) char           s_pool[POOL_SZ];   // ~113 KB
    __shared__ alignas(16) __hip_bfloat16 s_xin[Bsz][1024];  // 32 KB: [attended | mel]
    __shared__ alignas(16) __hip_bfloat16 s_h[Bsz][Hdim];    // 8 KB

    float*          s_q   = (float*)(s_pool + POOL_A);           // [16][512] f32
    __hip_bfloat16* s_gbf = (__hip_bfloat16*)(s_pool + POOL_A);  // [16][1024] bf16
    float*          s_sc  = (float*)(s_pool + POOL_B);           // scores
    __hip_bfloat16* s_ctx = (__hip_bfloat16*)(s_pool + POOL_C);  // [16][512] bf16

    int tid = threadIdx.x, lane = tid & 31, w = tid >> 5;
    __hip_bfloat16 bz = __float2bfloat16(0.0f);
    for (int i = tid; i < Bsz * MELD; i += 512)
        s_xin[i >> 9][512 + (i & 511)] = bz;                     // mel0 = 0
    __syncthreads();

    for (int step = 0; step < NSTEP; ++step) {
        // -- 1) q = mel @ Wq^T + bq  -> region A (f32) -------------------------
        for (int nt = w; nt < 32; nt += 16) {
            v8f acc = {};
            acc = mm_acc(&s_xin[0][512], 1024, 0, Wq, Edim, nt * 16, Edim, acc);
            int n = nt * 16 + (lane & 15);
            int mo = (lane >> 4) * 8;
            for (int r = 0; r < 8; ++r)
                s_q[(mo + r) * 512 + n] = acc[r] + bq[n];
        }
        __syncthreads();

        // -- 2a) scores + softmax: 2 threads per (b,h), wave32 shfl reduce -----
        if (tid < 256) {
            int row = tid >> 1, half = tid & 1;
            int b = row >> 3, h = row & 7;
            const float4* Kbh = (const float4*)(Kp + (size_t)row * Tlen * HDd);
            const float4* qv  = (const float4*)(s_q + b * 512 + h * HDd);
            int t0 = half * 64;
            float mx = -1e30f;
            for (int t = t0; t < t0 + 64; ++t) {
                const float4* kr = Kbh + t * (HDd / 4);
                float s = 0.0f;
#pragma unroll
                for (int c = 0; c < HDd / 4; ++c) {
                    float4 k4 = kr[c], q4 = qv[c];
                    s += k4.x * q4.x + k4.y * q4.y + k4.z * q4.z + k4.w * q4.w;
                }
                s *= 0.125f;                     // 1/sqrt(64)
                SC(s_sc, row, t) = s;
                mx = fmaxf(mx, s);
            }
            mx = fmaxf(mx, __shfl_xor(mx, 1, 32));
            float den = 0.0f;
            for (int t = t0; t < t0 + 64; ++t) {
                float e = __expf(SC(s_sc, row, t) - mx);
                SC(s_sc, row, t) = e;
                den += e;
            }
            den += __shfl_xor(den, 1, 32);
            float rd = 1.0f / den;
            for (int t = t0; t < t0 + 64; ++t) SC(s_sc, row, t) *= rd;
        } else if (step == 0) {
            // idle waves warm the cache hierarchy with the decoder weights
            const char* pw = (const char*)Wih;
            for (size_t off = (size_t)(tid - 256) * 512; off < (size_t)1024 * 1024 * 2;
                 off += 256 * 512)
                __builtin_prefetch(pw + off, 0, 1);
        }
        __syncthreads();

        // -- 2b) ctx = w @ V: thread pair splits the 64-d head dim -------------
        if (tid < 256) {
            int row = tid >> 1, half = tid & 1;
            int b = row >> 3, h = row & 7;
            const float4* Vbh = (const float4*)(Vp + (size_t)row * Tlen * HDd + half * 32);
            float acc[32];
#pragma unroll
            for (int j = 0; j < 32; ++j) acc[j] = 0.0f;
            for (int t = 0; t < Tlen; ++t) {
                float wt = SC(s_sc, row, t);
                const float4* vr = Vbh + t * (HDd / 4);
#pragma unroll
                for (int c = 0; c < 8; ++c) {
                    float4 v4 = vr[c];
                    acc[4 * c + 0] += wt * v4.x;
                    acc[4 * c + 1] += wt * v4.y;
                    acc[4 * c + 2] += wt * v4.z;
                    acc[4 * c + 3] += wt * v4.w;
                }
            }
#pragma unroll
            for (int j = 0; j < 32; ++j)
                s_ctx[b * 512 + h * HDd + half * 32 + j] = __float2bfloat16(acc[j]);
        }
        __syncthreads();

        // final step: head-averaged attention straight from the score buffer
        if (step == NSTEP - 1) {
            for (int i = tid; i < Bsz * Tlen; i += 512) {
                int b = i >> 7, t = i & 127;
                float a = 0.0f;
                for (int h = 0; h < NHead; ++h) a += SC(s_sc, b * 8 + h, t);
                attn_out[i] = a * 0.125f;
            }
        }

        // -- 3) attended = ctx @ Wo^T + bo -> s_xin[:,0:512] (bf16) ------------
        for (int nt = w; nt < 32; nt += 16) {
            v8f acc = {};
            acc = mm_acc(s_ctx, Edim, 0, Wo, Edim, nt * 16, Edim, acc);
            int n = nt * 16 + (lane & 15);
            int mo = (lane >> 4) * 8;
            for (int r = 0; r < 8; ++r)
                s_xin[mo + r][n] = __float2bfloat16(acc[r] + bo[n]);
        }
        __syncthreads();

        // -- 4) gates = xin @ dec_Wih^T + bih + bhh -> region A (bf16) ---------
        for (int nt = w; nt < 64; nt += 16) {
            v8f acc = {};
            acc = mm_acc(&s_xin[0][0], 1024, 0, Wih, 1024, nt * 16, 1024, acc);
            int n = nt * 16 + (lane & 15);
            float bb = bih[n] + bhh[n];
            int mo = (lane >> 4) * 8;
            for (int r = 0; r < 8; ++r)
                s_gbf[(mo + r) * 1024 + n] = __float2bfloat16(acc[r] + bb);
        }
        __syncthreads();

        // -- 5) h = sigmoid(o) * tanh(sigmoid(i) * tanh(g)) --------------------
        for (int i = tid; i < Bsz * Hdim; i += 512) {
            int b = i >> 8, j = i & (Hdim - 1);
            float ig = sigm(__bfloat162float(s_gbf[b * 1024 + j]));
            float gg = tanhfast(__bfloat162float(s_gbf[b * 1024 + 2 * Hdim + j]));
            float og = sigm(__bfloat162float(s_gbf[b * 1024 + 3 * Hdim + j]));
            float h  = og * tanhfast(ig * gg);
            s_h[b][j] = __float2bfloat16(h);
        }
        __syncthreads();

        // -- 6) mel = h @ Wmel^T + bmel -> global + feed back into s_xin -------
        for (int nt = w; nt < 32; nt += 16) {
            v8f acc = {};
            acc = mm_acc(&s_h[0][0], Hdim, 0, Wmel, Hdim, nt * 16, Hdim, acc);
            int n = nt * 16 + (lane & 15);
            float bb = bmel[n];
            int mo = (lane >> 4) * 8;
            for (int r = 0; r < 8; ++r) {
                int b = mo + r;
                float v = acc[r] + bb;
                mel_out[((size_t)b * NSTEP + step) * MELD + n] = v;
                s_xin[b][512 + n] = __float2bfloat16(v);
            }
        }
        // -- 7) stop = h @ Wstop^T + bstop -------------------------------------
        if (tid < Bsz) {
            float s = 0.0f;
            for (int j = 0; j < Hdim; ++j)
                s += __bfloat162float(s_h[tid][j]) * __bfloat162float(Wstop[j]);
            stop_out[tid * NSTEP + step] = s + bstop[0];
        }
        __syncthreads();
    }
}

// ---------------------------------------------------------------------------
// Host launcher
// ---------------------------------------------------------------------------
extern "C" void kernel_launch(void* const* d_in, const int* in_sizes, int n_in,
                              void* d_out, int out_size, void* d_ws, size_t ws_size,
                              hipStream_t stream) {
    (void)in_sizes; (void)n_in; (void)out_size; (void)ws_size;

    const int*   tok    = (const int*)d_in[0];
    const int*   spk    = (const int*)d_in[1];
    const float* etext  = (const float*)d_in[2];
    const float* espk   = (const float*)d_in[3];
    const float* fWih   = (const float*)d_in[4];
    const float* fWhh   = (const float*)d_in[5];
    const float* fbih   = (const float*)d_in[6];
    const float* fbhh   = (const float*)d_in[7];
    const float* bWih   = (const float*)d_in[8];
    const float* bWhh   = (const float*)d_in[9];
    const float* bbih   = (const float*)d_in[10];
    const float* bbhh   = (const float*)d_in[11];
    const float* Wq     = (const float*)d_in[12];
    const float* bq     = (const float*)d_in[13];
    const float* Wk     = (const float*)d_in[14];
    const float* bk     = (const float*)d_in[15];
    const float* Wv     = (const float*)d_in[16];
    const float* bv     = (const float*)d_in[17];
    const float* Wo     = (const float*)d_in[18];
    const float* bo     = (const float*)d_in[19];
    const float* dWih   = (const float*)d_in[20];
    /* d_in[21] = dec_Whh: unused (decoder state re-zeroed every step) */
    const float* dbih   = (const float*)d_in[22];
    const float* dbhh   = (const float*)d_in[23];
    const float* Wmel   = (const float*)d_in[24];
    const float* bmel   = (const float*)d_in[25];
    const float* Wstop  = (const float*)d_in[26];
    const float* bstop  = (const float*)d_in[27];

    // workspace carve (256B aligned)
    char* p = (char*)d_ws;
    auto alloc = [&](size_t bytes) -> void* {
        void* r = (void*)p;
        p += (bytes + 255) & ~(size_t)255;
        return r;
    };
    typedef __hip_bfloat16 bf;
    bf* wihF = (bf*)alloc((size_t)1024 * 256 * 2);
    bf* whhF = (bf*)alloc((size_t)1024 * 256 * 2);
    bf* wihB = (bf*)alloc((size_t)1024 * 256 * 2);
    bf* whhB = (bf*)alloc((size_t)1024 * 256 * 2);
    bf* wqb  = (bf*)alloc((size_t)512 * 512 * 2);
    bf* wkb  = (bf*)alloc((size_t)512 * 512 * 2);
    bf* wvb  = (bf*)alloc((size_t)512 * 512 * 2);
    bf* wob  = (bf*)alloc((size_t)512 * 512 * 2);
    bf* wdec = (bf*)alloc((size_t)1024 * 1024 * 2);
    bf* wmelb= (bf*)alloc((size_t)512 * 256 * 2);
    bf* wstpb= (bf*)alloc((size_t)256 * 2);
    bf* xbf  = (bf*)alloc((size_t)2048 * 256 * 2);
    float* xpreF = (float*)alloc((size_t)2048 * 1024 * 4);
    float* xpreB = (float*)alloc((size_t)2048 * 1024 * 4);
    bf* encbf = (bf*)alloc((size_t)2048 * 512 * 2);
    float* Kp = (float*)alloc((size_t)16 * 8 * 128 * 64 * 4);
    float* Vp = (float*)alloc((size_t)16 * 8 * 128 * 64 * 4);

    // K0: weight conversion
    struct { const float* s; bf* d; int n; } jobs[] = {
        {fWih, wihF, 1024 * 256}, {fWhh, whhF, 1024 * 256},
        {bWih, wihB, 1024 * 256}, {bWhh, whhB, 1024 * 256},
        {Wq, wqb, 512 * 512}, {Wk, wkb, 512 * 512},
        {Wv, wvb, 512 * 512}, {Wo, wob, 512 * 512},
        {dWih, wdec, 1024 * 1024}, {Wmel, wmelb, 512 * 256}, {Wstop, wstpb, 256},
    };
    for (auto& j : jobs)
        cvt_bf16_kernel<<<(j.n + 255) / 256, 256, 0, stream>>>(j.s, j.d, j.n);

    // K1: embedding
    embed_kernel<<<(Bsz * Tlen * Hdim) / 256, 256, 0, stream>>>(tok, spk, etext, espk, xbf);

    // K2: encoder input precompute x@Wih^T + bih + bhh (both directions)
    gemm_bias_f32_kernel<<<dim3(128, 8), 256, 0, stream>>>(xbf, wihF, fbih, fbhh,
                                                           xpreF, 256, 1024);
    gemm_bias_f32_kernel<<<dim3(128, 8), 256, 0, stream>>>(xbf, wihB, bbih, bbhh,
                                                           xpreB, 256, 1024);

    // K4: bidirectional LSTM (block 0 = forward, block 1 = backward)
    lstm_kernel<<<2, 512, 0, stream>>>(xpreF, xpreB, whhF, whhB, encbf);

    // K3: K/V projections, head-split f32
    gemm_kv_kernel<<<dim3(128, 4), 256, 0, stream>>>(encbf, wkb, bk, Kp);
    gemm_kv_kernel<<<dim3(128, 4), 256, 0, stream>>>(encbf, wvb, bv, Vp);

    // outputs: mel [16,1000,512] | stop [16,1000,1] | attn [16,1,128]
    float* mel_out  = (float*)d_out;
    float* stop_out = mel_out + (size_t)Bsz * NSTEP * MELD;
    float* attn_out = stop_out + (size_t)Bsz * NSTEP;

    // K5: 1000-step persistent decode
    decode_kernel<<<1, 512, 0, stream>>>(wqb, bq, wob, bo, wdec, dbih, dbhh,
                                         wmelb, bmel, wstpb, bstop,
                                         Kp, Vp, mel_out, stop_out, attn_out);
}
